// MultiHeadLatentAttention_57045755625537
// MI455X (gfx1250) — matrix operand order repaired and checked by reading, not compiled, not run on hardware
//
#include <hip/hip_runtime.h>
#include <cstdint>

typedef __attribute__((ext_vector_type(16))) __bf16 v16bf;
typedef __attribute__((ext_vector_type(8)))  __bf16 v8bf;
typedef __attribute__((ext_vector_type(8)))  float  v8f;
typedef __attribute__((ext_vector_type(4)))  unsigned int u32x4;
typedef __attribute__((ext_vector_type(4)))  int i32x4;
typedef __attribute__((ext_vector_type(8)))  int i32x8;

#define HIDDEN 2048
#define HEADS 16
#define HEAD_DIM 128
#define ROPE_DIM 64
#define LATENT 512
#define BB 4
#define SS 2048

#if defined(__has_builtin)
#if __has_builtin(__builtin_amdgcn_tensor_load_to_lds) && __has_builtin(__builtin_amdgcn_s_wait_tensorcnt)
#define HAVE_TDM 1
#endif
#endif
#ifndef HAVE_TDM
#define HAVE_TDM 0
#endif

union F16x16 { v16bf v; v8bf h[2]; };

#if HAVE_TDM
// 2D TDM tile load: `rows` rows of `row_elems` bf16, global row stride
// `row_stride_elems`, into LDS byte offset lds_off, with LDS padding
// (pad_interval: v -> 2^(v+1) DWORDs between pads; pad_amount: v -> v+1
// DWORDs inserted). D# packed per CDNA5 ISA 8.3/8.4; dims==tile (exact fit).
__device__ __forceinline__ void tdm_load_2d(uint32_t lds_off, const void* gptr,
                                            uint32_t row_elems, uint32_t rows,
                                            uint64_t row_stride_elems,
                                            int pad_interval, int pad_amount) {
  const uint64_t ga = (uint64_t)(uintptr_t)gptr;
  u32x4 g0;
  g0[0] = 1u;                                       // count=1, user desc
  g0[1] = lds_off;                                  // lds_addr
  g0[2] = (uint32_t)(ga & 0xFFFFFFFFu);             // global_addr[31:0]
  g0[3] = (uint32_t)((ga >> 32) & 0x01FFFFFFu) | (2u << 30);  // addr[56:32] | type=2
  i32x8 g1;
  uint32_t dw0 = (1u << 16);                        // data_size=1 (2 bytes)
  dw0 |= (1u << 20) | ((uint32_t)pad_interval << 22) | ((uint32_t)pad_amount << 25);
  g1[0] = (int)dw0;                                 // mask=0, pad cfg
  g1[1] = (int)((row_elems & 0xFFFFu) << 16);       // tensor_dim0[15:0] @ bit48
  g1[2] = (int)((row_elems >> 16) | ((rows & 0xFFFFu) << 16));   // dim0 hi | tensor_dim1 lo
  g1[3] = (int)(((rows >> 16) & 0xFFFFu) | ((row_elems & 0xFFFFu) << 16)); // dim1 hi | tile_dim0
  g1[4] = (int)(rows & 0xFFFFu);                    // tile_dim1 | tile_dim2=0
  g1[5] = (int)(uint32_t)(row_stride_elems & 0xFFFFFFFFu);       // stride0 lo
  g1[6] = (int)(uint32_t)((row_stride_elems >> 32) & 0xFFFFu);   // stride0 hi | stride1=0
  g1[7] = 0;
  i32x4 z4 = {0, 0, 0, 0};
#if __clang_major__ >= 23
  i32x8 z8 = {0, 0, 0, 0, 0, 0, 0, 0};
  __builtin_amdgcn_tensor_load_to_lds(g0, g1, z4, z4, z8, 0);
#else
  __builtin_amdgcn_tensor_load_to_lds(g0, g1, z4, z4, 0);
#endif
}
#endif  // HAVE_TDM

// ---------------------------------------------------------------- fp32 -> bf16
__global__ void f2bf_kernel(const float* __restrict__ in, __bf16* __restrict__ out, long n) {
  long i = (long)blockIdx.x * blockDim.x + threadIdx.x;
  long stride = (long)gridDim.x * blockDim.x;
  for (; i < n; i += stride) out[i] = (__bf16)in[i];
}

// fp32 [K,N] -> bf16 transposed [N,K], 32x32 LDS tiles, 256 threads
__global__ __launch_bounds__(256)
void f2bf_T_kernel(const float* __restrict__ in, __bf16* __restrict__ out, int K, int N) {
  __shared__ float t[32][33];
  const int k0 = blockIdx.y * 32, n0 = blockIdx.x * 32;
  const int tx = threadIdx.x & 31, ty = threadIdx.x >> 5;  // ty: 0..7
#pragma unroll
  for (int i = ty; i < 32; i += 8)
    t[i][tx] = in[(size_t)(k0 + i) * N + n0 + tx];
  __syncthreads();
#pragma unroll
  for (int i = ty; i < 32; i += 8)
    out[(size_t)(n0 + i) * K + k0 + tx] = (__bf16)t[tx][i];
}

// ---------------------------------------------------------------- tiled WMMA GEMM
// C[M,N] = A[M,K] (bf16 rm) @ W^T[N,K] (bf16, pre-transposed).
// Block tile 128x64, 8 waves of 32x32; K staged 64/round; TDM double-buffered.
// mode 0: bf16 rm out; mode 1: fp32 out + bias; mode 2: scatter V^T [B,H,128,S]
__global__ __launch_bounds__(256)
void gemm_bf16_kernel(const __bf16* __restrict__ A, const __bf16* __restrict__ WT,
                      __bf16* __restrict__ outb, float* __restrict__ outf,
                      const float* __restrict__ bias,
                      int M, int N, int K, int mode) {
  // padded stride 64 -> 72 halves (144 B)
  __shared__ alignas(16) __bf16 As[2][128 * 72];
  __shared__ alignas(16) __bf16 Bs[2][64 * 72];
  const int tid  = threadIdx.x;
  const int lane = tid & 31;
  const int wid  = tid >> 5;
  const int wm   = wid >> 1;          // 0..3 -> 32-row strip
  const int wn   = wid & 1;           // 0..1 -> 32-col strip
  const int mrow = lane & 15;
  const int hi   = (lane >> 4) & 1;
  const int m0 = blockIdx.y * 128;
  const int n0 = blockIdx.x * 64;

  v8f acc[2][2] = {};
  const int nstages = K >> 6;

#if HAVE_TDM
  const uint32_t asOff0 = (uint32_t)(uintptr_t)&As[0][0];
  const uint32_t asOff1 = (uint32_t)(uintptr_t)&As[1][0];
  const uint32_t bsOff0 = (uint32_t)(uintptr_t)&Bs[0][0];
  const uint32_t bsOff1 = (uint32_t)(uintptr_t)&Bs[1][0];
  if (wid == 0) {  // prologue: stages 0,1 -> buffers 0,1
    tdm_load_2d(asOff0, A + (size_t)m0 * K, 64, 128, (uint64_t)K, 4, 3);
    tdm_load_2d(bsOff0, WT + (size_t)n0 * K, 64, 64, (uint64_t)K, 4, 3);
    if (nstages > 1) {
      tdm_load_2d(asOff1, A + (size_t)m0 * K + 64, 64, 128, (uint64_t)K, 4, 3);
      tdm_load_2d(bsOff1, WT + (size_t)n0 * K + 64, 64, 64, (uint64_t)K, 4, 3);
    }
  }
#endif

  for (int s = 0; s < nstages; ++s) {
    const int cur = s & 1;
#if HAVE_TDM
    if (wid == 0) {  // retire stage s: if stage s+1 is in flight, <=2 suffices
      if (s + 1 < nstages) __builtin_amdgcn_s_wait_tensorcnt(2);
      else                 __builtin_amdgcn_s_wait_tensorcnt(0);
    }
    __syncthreads();                                     // publish buf[cur]
#else
    __syncthreads();
    {  // cooperative stage: A 128x64, B 64x64 (rows contiguous)
      const int k0 = s * 64;
      const int ar = tid >> 1, ac = (tid & 1) * 32;      // A: 32 halves/thread
#pragma unroll
      for (int j = 0; j < 4; ++j)
        *(v8bf*)(&As[cur][ar * 72 + ac + j * 8]) =
            *(const v8bf*)(A + (size_t)(m0 + ar) * K + k0 + ac + j * 8);
      const int br = tid >> 2, bc = (tid & 3) * 16;      // B: 16 halves/thread
#pragma unroll
      for (int j = 0; j < 2; ++j)
        *(v8bf*)(&Bs[cur][br * 72 + bc + j * 8]) =
            *(const v8bf*)(WT + (size_t)(n0 + br) * K + k0 + bc + j * 8);
    }
    __syncthreads();
#endif
    // 8 WMMAs per wave per stage
#pragma unroll
    for (int c = 0; c < 2; ++c) {
      F16x16 af[2], bf_[2];
#pragma unroll
      for (int mt = 0; mt < 2; ++mt) {
        const __bf16* ap = &As[cur][(wm * 32 + mt * 16 + mrow) * 72 + c * 32 + hi * 8];
        af[mt].h[0] = *(const v8bf*)ap;
        af[mt].h[1] = *(const v8bf*)(ap + 16);
      }
#pragma unroll
      for (int nt = 0; nt < 2; ++nt) {
        const __bf16* bp = &Bs[cur][(wn * 32 + nt * 16 + mrow) * 72 + c * 32 + hi * 8];
        bf_[nt].h[0] = *(const v8bf*)bp;
        bf_[nt].h[1] = *(const v8bf*)(bp + 16);
      }
#pragma unroll
      for (int mt = 0; mt < 2; ++mt)
#pragma unroll
        for (int nt = 0; nt < 2; ++nt)
          acc[mt][nt] = __builtin_amdgcn_wmma_f32_16x16x32_bf16(
              false, af[mt].v, false, bf_[nt].v, (short)0, acc[mt][nt], false, false);
    }
#if HAVE_TDM
    __syncthreads();  // buf[cur] reads done -> reusable
    if (wid == 0 && s + 2 < nstages) {
      const int k2 = (s + 2) * 64;
      tdm_load_2d(cur ? asOff1 : asOff0, A + (size_t)m0 * K + k2, 64, 128, (uint64_t)K, 4, 3);
      tdm_load_2d(cur ? bsOff1 : bsOff0, WT + (size_t)n0 * K + k2, 64, 64, (uint64_t)K, 4, 3);
    }
#endif
  }

#pragma unroll
  for (int mt = 0; mt < 2; ++mt)
#pragma unroll
    for (int nt = 0; nt < 2; ++nt)
#pragma unroll
      for (int r = 0; r < 8; ++r) {
        int m = m0 + wm * 32 + mt * 16 + hi * 8 + r;
        int n = n0 + wn * 32 + nt * 16 + mrow;
        float c = acc[mt][nt][r];
        if (mode == 0) {
          outb[(size_t)m * N + n] = (__bf16)c;
        } else if (mode == 1) {
          outf[(size_t)m * N + n] = c + bias[n];
        } else {  // scatter v TRANSPOSED into [B,H,128,S]
          int b = m >> 11, sIdx = m & (SS - 1);
          int h = n >> 7, d = n & (HEAD_DIM - 1);
          outb[(((size_t)(b * HEADS + h)) * HEAD_DIM + d) * SS + sIdx] = (__bf16)c;
        }
      }
}

// ---------------------------------------------------------------- RoPE + layout
__global__ void assemble_q_kernel(const __bf16* __restrict__ qc, const __bf16* __restrict__ qr,
                                  __bf16* __restrict__ q) {
  int idx = blockIdx.x * blockDim.x + threadIdx.x;  // B*S*H*64
  int d = idx & 63;
  int h = (idx >> 6) & (HEADS - 1);
  int s = (idx >> 10) & (SS - 1);
  int b = idx >> 21;
  size_t bs = (size_t)b * SS + s;
  int j = d & 31;
  float inv = __powf(10000.f, -(float)(2 * j) / (float)ROPE_DIM);
  float ang = (float)s * inv;
  float cv = __cosf(ang), sv = __sinf(ang);
  float xr = (float)qr[bs * 1024 + h * 64 + d];
  float xo = (d < 32) ? -(float)qr[bs * 1024 + h * 64 + d + 32]
                      :  (float)qr[bs * 1024 + h * 64 + d - 32];
  size_t o = (((size_t)(b * HEADS + h)) * SS + s) * HEAD_DIM;
  q[o + d]      = qc[bs * 1024 + h * 64 + d];
  q[o + 64 + d] = (__bf16)(xr * cv + xo * sv);
}

__global__ void assemble_k_kernel(const __bf16* __restrict__ kc, const __bf16* __restrict__ krr,
                                  __bf16* __restrict__ k) {
  int idx = blockIdx.x * blockDim.x + threadIdx.x;
  int d = idx & 63;
  int h = (idx >> 6) & (HEADS - 1);
  int s = (idx >> 10) & (SS - 1);
  int b = idx >> 21;
  size_t bs = (size_t)b * SS + s;
  int j = d & 31;
  float inv = __powf(10000.f, -(float)(2 * j) / (float)ROPE_DIM);
  float ang = (float)s * inv;
  float cv = __cosf(ang), sv = __sinf(ang);
  float xr = (float)krr[bs * 64 + d];
  float xo = (d < 32) ? -(float)krr[bs * 64 + d + 32]
                      :  (float)krr[bs * 64 + d - 32];
  size_t o = (((size_t)(b * HEADS + h)) * SS + s) * HEAD_DIM;
  k[o + d]      = kc[bs * 1024 + h * 64 + d];
  k[o + 64 + d] = (__bf16)(xr * cv + xo * sv);
}

// ---------------------------------------------------------------- flash attention
// block = (q-tile of 64, head, batch); 4 waves x 16 q-rows each.
// K [B,H,S,128] and V^T [B,H,128,S] tiles are TDM-staged, double-buffered on
// TENSORcnt (2 descriptors per stage), with padded LDS rows (no bank conflicts).
__global__ __launch_bounds__(128)
void mla_attn_kernel(const __bf16* __restrict__ Q, const __bf16* __restrict__ Kt,
                     const __bf16* __restrict__ Vt, __bf16* __restrict__ ctx) {
  __shared__ alignas(16) __bf16 Ks[2][64 * 136];    // K tile [key][dim], pad 128->136
  __shared__ alignas(16) __bf16 vTs[2][128 * 72];   // V^T tile [dim][key], pad 64->72
  __shared__ alignas(16) __bf16 Ps[4 * 16 * 72];    // per-wave P staging [row][key]
  const int tid  = threadIdx.x;
  const int lane = tid & 31;
  const int wid  = tid >> 5;
  const int mrow = lane & 15;
  const int hi   = (lane >> 4) & 1;
  const int b = blockIdx.z, h = blockIdx.y;
  const int q0 = blockIdx.x * 64;
  const size_t head_off = ((size_t)(b * HEADS + h)) * SS * HEAD_DIM;
  const __bf16* qb = Q + head_off;
  const __bf16* kb = Kt + head_off;
  const __bf16* vb = Vt + head_off;   // [128][S] within head

  // Q fragments (A-layout), registers for all k-tiles
  F16x16 qf[4];
  {
    const __bf16* qrow = qb + (size_t)(q0 + wid * 16 + mrow) * HEAD_DIM;
#pragma unroll
    for (int c = 0; c < 4; ++c) {
      const __bf16* p = qrow + c * 32 + hi * 8;
      qf[c].h[0] = *(const v8bf*)p;
      qf[c].h[1] = *(const v8bf*)(p + 16);
    }
  }

  float mst[8], lst[8];
#pragma unroll
  for (int r = 0; r < 8; ++r) { mst[r] = -3.0e38f; lst[r] = 0.f; }
  v8f oacc[8] = {};

  const int ktiles = (q0 >> 6) + 1;  // causal: tiles 0 .. q0/64

#if HAVE_TDM
  const uint32_t ksOff0 = (uint32_t)(uintptr_t)&Ks[0][0];
  const uint32_t ksOff1 = (uint32_t)(uintptr_t)&Ks[1][0];
  const uint32_t vsOff0 = (uint32_t)(uintptr_t)&vTs[0][0];
  const uint32_t vsOff1 = (uint32_t)(uintptr_t)&vTs[1][0];
  if (wid == 0) {  // prologue: stages 0,1
    tdm_load_2d(ksOff0, kb, 128, 64, (uint64_t)HEAD_DIM, 5, 3);
    tdm_load_2d(vsOff0, vb, 64, 128, (uint64_t)SS, 4, 3);
    if (ktiles > 1) {
      tdm_load_2d(ksOff1, kb + (size_t)64 * HEAD_DIM, 128, 64, (uint64_t)HEAD_DIM, 5, 3);
      tdm_load_2d(vsOff1, vb + 64, 64, 128, (uint64_t)SS, 4, 3);
    }
  }
#endif

  for (int kt = 0; kt < ktiles; ++kt) {
    const int k0 = kt * 64;
    const int cur = kt & 1;
#if HAVE_TDM
    if (wid == 0) {
      if (kt + 1 < ktiles) __builtin_amdgcn_s_wait_tensorcnt(2);
      else                 __builtin_amdgcn_s_wait_tensorcnt(0);
    }
    __syncthreads();  // buf[cur] published; prior Ps readers done
#else
    __syncthreads();
    {  // K tile: 64 rows x 128 halves
      const int key = tid >> 1, dbase = (tid & 1) * 64;
      const __bf16* ksrc = kb + (size_t)(k0 + key) * HEAD_DIM + dbase;
#pragma unroll
      for (int j = 0; j < 8; ++j)
        *(v8bf*)(&Ks[cur][key * 136 + dbase + j * 8]) = ((const v8bf*)ksrc)[j];
    }
    {  // V^T tile: 128 rows x 64 halves (V already transposed in global)
      const __bf16* vsrc = vb + (size_t)tid * SS + k0;
#pragma unroll
      for (int j = 0; j < 8; ++j)
        *(v8bf*)(&vTs[cur][tid * 72 + j * 8]) = ((const v8bf*)vsrc)[j];
    }
    __syncthreads();
#endif

    // scores: S[16x64] = Q[16x128] @ K^T, B-fragments from LDS K rows
    v8f sacc[4] = {};
#pragma unroll
    for (int c = 0; c < 4; ++c) {
#pragma unroll
      for (int nt = 0; nt < 4; ++nt) {
        const __bf16* kp = &Ks[cur][(nt * 16 + mrow) * 136 + c * 32 + hi * 8];
        F16x16 bf_;
        bf_.h[0] = *(const v8bf*)kp;
        bf_.h[1] = *(const v8bf*)(kp + 16);
        sacc[nt] = __builtin_amdgcn_wmma_f32_16x16x32_bf16(false, qf[c].v, false, bf_.v,
                                                           (short)0, sacc[nt], false, false);
      }
    }

    // scale + causal mask (-10000, as in reference) + online softmax
    const float scale = 0.08838834764831845f;  // 1/sqrt(128)
#pragma unroll
    for (int r = 0; r < 8; ++r) {
      const int qi = q0 + wid * 16 + hi * 8 + r;
      float rmax = -3.0e38f;
#pragma unroll
      for (int nt = 0; nt < 4; ++nt) {
        const int kj = k0 + nt * 16 + mrow;
        float sv = sacc[nt][r] * scale;
        if (kj > qi) sv = -10000.f;
        sacc[nt][r] = sv;
        rmax = fmaxf(rmax, sv);
      }
#pragma unroll
      for (int off = 1; off < 16; off <<= 1)
        rmax = fmaxf(rmax, __shfl_xor(rmax, off, 32));
      const float mnew  = fmaxf(mst[r], rmax);
      const float alpha = __expf(mst[r] - mnew);
      mst[r] = mnew;
      lst[r] *= alpha;
#pragma unroll
      for (int d = 0; d < 8; ++d) oacc[d][r] *= alpha;
      float rsum = 0.f;
#pragma unroll
      for (int nt = 0; nt < 4; ++nt) {
        float p = __expf(sacc[nt][r] - mnew);
        sacc[nt][r] = p;
        rsum += p;
      }
#pragma unroll
      for (int off = 1; off < 16; off <<= 1)
        rsum += __shfl_xor(rsum, off, 32);
      lst[r] += rsum;
    }

    // stage P (bf16) C-layout -> LDS, re-read in A-layout
    __bf16* pw = Ps + wid * 16 * 72;
#pragma unroll
    for (int r = 0; r < 8; ++r)
#pragma unroll
      for (int nt = 0; nt < 4; ++nt)
        pw[(hi * 8 + r) * 72 + nt * 16 + mrow] = (__bf16)sacc[nt][r];
    __syncthreads();  // publish P

    // O[16x128] += P[16x64] @ V[64x128]
#pragma unroll
    for (int c2 = 0; c2 < 2; ++c2) {
      F16x16 pa;
      const __bf16* pp = pw + mrow * 72 + c2 * 32 + hi * 8;
      pa.h[0] = *(const v8bf*)pp;
      pa.h[1] = *(const v8bf*)(pp + 16);
#pragma unroll
      for (int d = 0; d < 8; ++d) {
        const __bf16* vp = &vTs[cur][(d * 16 + mrow) * 72 + c2 * 32 + hi * 8];
        F16x16 bv;
        bv.h[0] = *(const v8bf*)vp;
        bv.h[1] = *(const v8bf*)(vp + 16);
        oacc[d] = __builtin_amdgcn_wmma_f32_16x16x32_bf16(false, pa.v, false, bv.v,
                                                          (short)0, oacc[d], false, false);
      }
    }
#if HAVE_TDM
    __syncthreads();  // all reads of buf[cur] + Ps done -> reusable
    if (wid == 0 && kt + 2 < ktiles) {
      const int k2 = (kt + 2) * 64;
      tdm_load_2d(cur ? ksOff1 : ksOff0, kb + (size_t)k2 * HEAD_DIM, 128, 64,
                  (uint64_t)HEAD_DIM, 5, 3);
      tdm_load_2d(cur ? vsOff1 : vsOff0, vb + k2, 64, 128, (uint64_t)SS, 4, 3);
    }
#else
    __syncthreads();
#endif
  }

  // normalize, write ctx as [B, S, H*128] bf16 (feeds final GEMM)
#pragma unroll
  for (int r = 0; r < 8; ++r) {
    const int qi = q0 + wid * 16 + hi * 8 + r;
    const float inv = 1.0f / lst[r];
    size_t rowo = ((size_t)b * SS + qi) * HIDDEN + (size_t)h * HEAD_DIM;
#pragma unroll
    for (int d = 0; d < 8; ++d)
      ctx[rowo + d * 16 + mrow] = (__bf16)(oacc[d][r] * inv);
  }
}

// ---------------------------------------------------------------- launch
extern "C" void kernel_launch(void* const* d_in, const int* in_sizes, int n_in,
                              void* d_out, int out_size, void* d_ws, size_t ws_size,
                              hipStream_t stream) {
  (void)in_sizes; (void)n_in; (void)out_size; (void)ws_size;
  const float* x        = (const float*)d_in[0];
  const float* wq_down  = (const float*)d_in[1];
  const float* wq_up    = (const float*)d_in[2];
  const float* wq_rope  = (const float*)d_in[3];
  const float* wk_rope  = (const float*)d_in[4];
  const float* wkv_down = (const float*)d_in[5];
  const float* wk_up    = (const float*)d_in[6];
  const float* wv_up    = (const float*)d_in[7];
  const float* wo       = (const float*)d_in[8];
  const float* bo       = (const float*)d_in[9];

  size_t off = 0;
  auto carve = [&](size_t elems) -> __bf16* {
    __bf16* p = (__bf16*)((char*)d_ws + off);
    off = (off + elems * 2 + 255) & ~(size_t)255;
    return p;
  };
  const size_t MS = (size_t)BB * SS;  // 8192
  __bf16* xb     = carve(MS * HIDDEN);
  __bf16* wqd_t  = carve((size_t)HIDDEN * LATENT);   // [512][2048]
  __bf16* wqu_t  = carve((size_t)LATENT * 1024);     // [1024][512]
  __bf16* wqr_t  = carve((size_t)LATENT * 1024);     // [1024][512]
  __bf16* wkr_t  = carve((size_t)HIDDEN * 64);       // [64][2048]
  __bf16* wkvd_t = carve((size_t)HIDDEN * LATENT);   // [512][2048]
  __bf16* wku_t  = carve((size_t)LATENT * 1024);     // [1024][512]
  __bf16* wvu_t  = carve((size_t)LATENT * HIDDEN);   // [2048][512]
  __bf16* wo_t   = carve((size_t)HIDDEN * HIDDEN);   // [2048][2048]
  __bf16* qlat   = carve(MS * LATENT);
  __bf16* kvlat  = carve(MS * LATENT);
  __bf16* qc     = carve(MS * 1024);
  __bf16* qr     = carve(MS * 1024);
  __bf16* kc     = carve(MS * 1024);
  __bf16* krr    = carve(MS * 64);
  __bf16* qT     = carve(MS * HIDDEN);
  __bf16* kT     = carve(MS * HIDDEN);
  __bf16* vT     = carve(MS * HIDDEN);               // [B,H,128,S] (transposed)
  __bf16* ctx    = carve(MS * HIDDEN);

  f2bf_kernel<<<(int)((MS * HIDDEN + 255) / 256), 256, 0, stream>>>(x, xb, (long)MS * HIDDEN);
  auto cvtT = [&](const float* src, __bf16* dst, int K, int N) {
    f2bf_T_kernel<<<dim3(N / 32, K / 32), 256, 0, stream>>>(src, dst, K, N);
  };
  cvtT(wq_down,  wqd_t,  HIDDEN, LATENT);
  cvtT(wq_up,    wqu_t,  LATENT, 1024);
  cvtT(wq_rope,  wqr_t,  LATENT, 1024);
  cvtT(wk_rope,  wkr_t,  HIDDEN, 64);
  cvtT(wkv_down, wkvd_t, HIDDEN, LATENT);
  cvtT(wk_up,    wku_t,  LATENT, 1024);
  cvtT(wv_up,    wvu_t,  LATENT, HIDDEN);
  cvtT(wo,       wo_t,   HIDDEN, HIDDEN);

  auto gemm = [&](const __bf16* A, const __bf16* WT, __bf16* ob, float* of,
                  const float* bias, int M, int N, int K, int mode) {
    dim3 g(N / 64, M / 128);
    gemm_bf16_kernel<<<g, 256, 0, stream>>>(A, WT, ob, of, bias, M, N, K, mode);
  };
  gemm(xb,    wqd_t,  qlat,  nullptr, nullptr, (int)MS, LATENT, HIDDEN, 0);
  gemm(xb,    wkvd_t, kvlat, nullptr, nullptr, (int)MS, LATENT, HIDDEN, 0);
  gemm(xb,    wkr_t,  krr,   nullptr, nullptr, (int)MS, 64,     HIDDEN, 0);
  gemm(qlat,  wqu_t,  qc,    nullptr, nullptr, (int)MS, 1024,   LATENT, 0);
  gemm(qlat,  wqr_t,  qr,    nullptr, nullptr, (int)MS, 1024,   LATENT, 0);
  gemm(kvlat, wku_t,  kc,    nullptr, nullptr, (int)MS, 1024,   LATENT, 0);
  gemm(kvlat, wvu_t,  vT,    nullptr, nullptr, (int)MS, HIDDEN, LATENT, 2);

  const int an = BB * SS * HEADS * 64;
  assemble_q_kernel<<<an / 256, 256, 0, stream>>>(qc, qr, qT);
  assemble_k_kernel<<<an / 256, 256, 0, stream>>>(kc, krr, kT);

  mla_attn_kernel<<<dim3(SS / 64, HEADS, BB), 128, 0, stream>>>(qT, kT, vT, ctx);

  gemm(ctx, wo_t, nullptr, (float*)d_out, bo, (int)MS, HIDDEN, HIDDEN, 1);
}
